// autoencoder_rnn1_15848429322981
// MI455X (gfx1250) — matrix operand order repaired and checked by reading, not compiled, not run on hardware
//
#include <hip/hip_runtime.h>
#include <hip/hip_bf16.h>

// ---------------------------------------------------------------------------
// LSTM autoencoder on MI455X (gfx1250).
// Strategy: weights -> bf16, pre-packed into V_WMMA_F32_16X16X32_BF16
// A-fragment layout; persistent kernel with software grid barrier walks the
// 4096 encoder + 4094 decoder steps; all per-step weight traffic hits the
// 192MB L2 (total packed weights ~75MB). Decoder uses (W_ih + W_hh) for t>=1
// since prev==h in the feedback loop. Per-step input vector is staged into
// LDS with GLOBAL_LOAD_ASYNC_TO_LDS_B128 (ASYNCcnt path, no VGPR round trip).
// ---------------------------------------------------------------------------

typedef __attribute__((ext_vector_type(16))) __bf16 v16bf;
typedef __attribute__((ext_vector_type(8)))  float  v8f;

#define L_SEQ   4096
#define E_DIM   1024
#define H_DIM   2048
#define LAT_DIM 768
#define NWG     128      // persistent workgroups (also gridDim)
#define NT_ENC  128      // encoder output tiles: 2048/16
#define KC_ENC  96       // encoder K chunks: (1024+2048)/32
#define NT_LS   48       // latent tiles: 768/16
#define KC_LS   64       // 2048/32
#define NT_DEC  64       // decoder tiles: 1024/16
#define KC_DEC0 64       // step-0 K chunks: (1024+1024)/32
#define KC_DECS 32       // steady-state K chunks: 1024/32

static __device__ __forceinline__ unsigned short f2bf_rne(float f) {
  unsigned u = __float_as_uint(f);
  u += 0x7FFFu + ((u >> 16) & 1u);
  return (unsigned short)(u >> 16);
}

static __device__ __forceinline__ float sigm(float x) {
  return 1.0f / (1.0f + expf(-x));
}

// (lane, e) -> K index inside a 16-bit A fragment 16x32 (contiguous, matches B):
// lanes 0-15 hold K {0..15}; lanes 16-31 hold K {16..31}.
static __device__ __forceinline__ int a_kmap(int lane, int e) {
  return (lane & 16) + e;
}

// ---- async memory->LDS staging (CDNA5 GLOBAL_LOAD_ASYNC_TO_LDS_B128) ------
// LDS destination address = low 32 bits of the flat pointer (HW truncates
// flat addresses into LDS space, ISA 10.2); tracked by ASYNCcnt.

static __device__ __forceinline__ void async_b128_to_lds(void* lds_dst, const void* gsrc) {
  unsigned ldsa = (unsigned)(unsigned long long)(size_t)lds_dst;
  unsigned long long ga = (unsigned long long)(size_t)gsrc;
  asm volatile("global_load_async_to_lds_b128 %0, %1, off"
               :: "v"(ldsa), "v"(ga)
               : "memory");
}

static __device__ __forceinline__ void wait_async0() {
  asm volatile("s_wait_asynccnt 0" ::: "memory");
}

// copy `bytes` (multiple of 16) from global to LDS, 16B per thread per pass
static __device__ __forceinline__ void stage_async(void* lds, const void* g,
                                                   int bytes, int tid) {
  for (int o = tid * 16; o < bytes; o += 256 * 16)
    async_b128_to_lds((char*)lds + o, (const char*)g + o);
}

// ------------------------- phase 0: weight packing -------------------------

__global__ void k_pack_enc(const float* __restrict__ Wih, const float* __restrict__ Whh,
                           unsigned short* __restrict__ A) {
  size_t i = (size_t)blockIdx.x * 256 + threadIdx.x;
  if (i >= (size_t)4 * NT_ENC * KC_ENC * 512) return;
  int e = (int)(i & 15);
  size_t t = i >> 4;
  int lane = (int)(t & 31);
  size_t f = t >> 5;
  int kc = (int)(f % KC_ENC);
  size_t f2 = f / KC_ENC;
  int jt = (int)(f2 % NT_ENC);
  int g  = (int)(f2 / NT_ENC);
  int row = g * H_DIM + jt * 16 + (lane & 15);
  int col = kc * 32 + a_kmap(lane, e);
  float w = (col < E_DIM) ? Wih[(size_t)row * E_DIM + col]
                          : Whh[(size_t)row * H_DIM + (col - E_DIM)];
  A[i] = f2bf_rne(w);
}

__global__ void k_pack_ls(const float* __restrict__ Wls, unsigned short* __restrict__ A) {
  size_t i = (size_t)blockIdx.x * 256 + threadIdx.x;
  if (i >= (size_t)NT_LS * KC_LS * 512) return;
  int e = (int)(i & 15);
  size_t t = i >> 4;
  int lane = (int)(t & 31);
  size_t f = t >> 5;
  int kc = (int)(f % KC_LS);
  int jt = (int)(f / KC_LS);
  int row = jt * 16 + (lane & 15);
  int col = kc * 32 + a_kmap(lane, e);
  A[i] = f2bf_rne(Wls[(size_t)row * H_DIM + col]);
}

__global__ void k_pack_dec0(const float* __restrict__ Wih, const float* __restrict__ Whh,
                            unsigned short* __restrict__ A) {
  size_t i = (size_t)blockIdx.x * 256 + threadIdx.x;
  if (i >= (size_t)4 * NT_DEC * KC_DEC0 * 512) return;
  int e = (int)(i & 15);
  size_t t = i >> 4;
  int lane = (int)(t & 31);
  size_t f = t >> 5;
  int kc = (int)(f % KC_DEC0);
  size_t f2 = f / KC_DEC0;
  int jt = (int)(f2 % NT_DEC);
  int g  = (int)(f2 / NT_DEC);
  int row = g * E_DIM + jt * 16 + (lane & 15);
  int col = kc * 32 + a_kmap(lane, e);
  float w = (col < E_DIM) ? Wih[(size_t)row * E_DIM + col]
                          : Whh[(size_t)row * E_DIM + (col - E_DIM)];
  A[i] = f2bf_rne(w);
}

__global__ void k_pack_decS(const float* __restrict__ Wih, const float* __restrict__ Whh,
                            unsigned short* __restrict__ A) {
  size_t i = (size_t)blockIdx.x * 256 + threadIdx.x;
  if (i >= (size_t)4 * NT_DEC * KC_DECS * 512) return;
  int e = (int)(i & 15);
  size_t t = i >> 4;
  int lane = (int)(t & 31);
  size_t f = t >> 5;
  int kc = (int)(f % KC_DECS);
  size_t f2 = f / KC_DECS;
  int jt = (int)(f2 % NT_DEC);
  int g  = (int)(f2 / NT_DEC);
  int row = g * E_DIM + jt * 16 + (lane & 15);
  int col = kc * 32 + a_kmap(lane, e);
  A[i] = f2bf_rne(Wih[(size_t)row * E_DIM + col] + Whh[(size_t)row * E_DIM + col]);
}

__global__ void k_misc(const float* __restrict__ bih_e, const float* __restrict__ bhh_e,
                       const float* __restrict__ bih_d, const float* __restrict__ bhh_d,
                       const float* __restrict__ emb, const float* __restrict__ embV,
                       const float* __restrict__ embJ,
                       const int* __restrict__ Vg, const int* __restrict__ Jg,
                       float* b_enc, float* b_dec, unsigned short* h_enc,
                       unsigned short* h_dec, unsigned short* prev0,
                       unsigned* bar, float* out) {
  int i = blockIdx.x * 256 + threadIdx.x;
  if (i < 4 * H_DIM) b_enc[i] = bih_e[i] + bhh_e[i];
  if (i < 4 * E_DIM) b_dec[i] = bih_d[i] + bhh_d[i];
  if (i < 2 * H_DIM) h_enc[i] = 0;            // encoder h0 = 0 (both buffers)
  if (i < 2) bar[i] = 0;                      // grid-barrier counter + generation
  if (i < E_DIM) {
    float xv = emb[23 * E_DIM + i];           // X_IDX row
    prev0[i] = f2bf_rne(xv);
    out[(size_t)E_DIM + (size_t)(L_SEQ - 1) * E_DIM + i] = xv;   // recon[L-1]=emb[X]
    out[(size_t)E_DIM + i] = emb[1 * E_DIM + i];                 // recon[0]=emb[B]
  }
  if (i < 128) {
    float v = embV[(size_t)Vg[0] * 128 + i];
    out[i] = v;             h_dec[i] = f2bf_rne(v);
    float jv = embJ[(size_t)Jg[0] * 128 + i];
    out[896 + i] = jv;      h_dec[896 + i] = f2bf_rne(jv);
  }
}

__global__ void k_embseq(const int* __restrict__ x, const float* __restrict__ emb,
                         unsigned short* __restrict__ eseq, float* out) {
  size_t i = (size_t)blockIdx.x * 256 + threadIdx.x;
  if (i >= (size_t)L_SEQ * E_DIM) return;
  int t = (int)(i >> 10), col = (int)(i & 1023);
  float v = emb[(size_t)x[t] * E_DIM + col];
  eseq[i] = f2bf_rne(v);
  out[(size_t)E_DIM + (size_t)L_SEQ * E_DIM + i] = v;  // 3rd output: embedded
}

// ------------------------- software grid barrier ---------------------------

static __device__ void gsync(unsigned* bar) {
  __threadfence();
  __syncthreads();
  if (threadIdx.x == 0) {
    unsigned gen = __atomic_load_n(&bar[1], __ATOMIC_ACQUIRE);
    if (atomicAdd(&bar[0], 1u) == NWG - 1u) {
      bar[0] = 0u;
      __atomic_store_n(&bar[1], gen + 1u, __ATOMIC_RELEASE);
    } else {
      while (__atomic_load_n(&bar[1], __ATOMIC_ACQUIRE) == gen)
        __builtin_amdgcn_s_sleep(2);
    }
  }
  __syncthreads();
}

// ------------------------- persistent LSTM kernel --------------------------

__global__ __launch_bounds__(256, 1)
void k_lstm_persistent(const v16bf* __restrict__ Aenc, const v16bf* __restrict__ Als,
                       const v16bf* __restrict__ Adec0, const v16bf* __restrict__ AdecS,
                       const unsigned short* __restrict__ eseq,
                       const float* __restrict__ b_enc, const float* __restrict__ b_dec,
                       const float* __restrict__ b_ls,
                       unsigned short* h_enc, unsigned short* h_dec,
                       const unsigned short* __restrict__ prev0,
                       unsigned* bar, float* out) {
  __shared__ __align__(64) unsigned short vls[E_DIM + H_DIM];  // input vector (bf16)
  __shared__ float red[8][16];                                 // cross-wave partials
  __shared__ float cst[16];                                    // persistent cell state

  const int wg   = blockIdx.x;
  const int tid  = threadIdx.x;
  const int wv   = tid >> 5;
  const int lane = tid & 31;
  const int gate = wv >> 1;            // 2 waves per gate
  const int half = wv & 1;             // K-range split
  const int boff = (lane & 16) ? 16 : 0;  // B fragment: lanes 0-15 K[0:16), 16-31 K[16:32)

  if (tid < 16) cst[tid] = 0.0f;

  // ======================= encoder: 4096 steps ===========================
  for (int t = 0; t < L_SEQ; ++t) {
    stage_async(vls, eseq + (size_t)t * E_DIM, E_DIM * 2, tid);
    stage_async(vls + E_DIM, h_enc + (size_t)(t & 1) * H_DIM, H_DIM * 2, tid);
    wait_async0();
    __syncthreads();

    v8f acc = {};
    const size_t fb = ((size_t)gate * NT_ENC + wg) * KC_ENC;
    const int kc0 = half * (KC_ENC / 2), kc1 = kc0 + KC_ENC / 2;
    for (int kc = kc0; kc < kc1; ++kc) {
      __builtin_prefetch((const void*)(Aenc + (fb + kc + 8) * 32 + lane), 0, 3);
      v16bf a = Aenc[(fb + kc) * 32 + lane];
      v16bf b = *(const v16bf*)&vls[kc * 32 + boff];
      acc = __builtin_amdgcn_wmma_f32_16x16x32_bf16(false, a, false, b,
                                                    (short)0, acc, false, false);
    }
    if (lane == 0)       { for (int r = 0; r < 8; ++r) red[wv][r]     = acc[r]; }
    else if (lane == 16) { for (int r = 0; r < 8; ++r) red[wv][8 + r] = acc[r]; }
    __syncthreads();

    if (tid < 16) {
      const int jg = wg * 16 + tid;
      float gi = red[0][tid] + red[1][tid] + b_enc[0 * H_DIM + jg];
      float gf = red[2][tid] + red[3][tid] + b_enc[1 * H_DIM + jg];
      float gg = red[4][tid] + red[5][tid] + b_enc[2 * H_DIM + jg];
      float go = red[6][tid] + red[7][tid] + b_enc[3 * H_DIM + jg];
      float c = sigm(gf) * cst[tid] + sigm(gi) * tanhf(gg);
      cst[tid] = c;
      h_enc[(size_t)((t + 1) & 1) * H_DIM + jg] = f2bf_rne(sigm(go) * tanhf(c));
    }
    gsync(bar);
  }

  // ======================= latent projection =============================
  if (wg < NT_LS) {
    stage_async(vls, h_enc, H_DIM * 2, tid);   // buffer 0 = h_n
    wait_async0();
    __syncthreads();
    v8f acc = {};
    const size_t fb = (size_t)wg * KC_LS;
    for (int kc = wv * 8; kc < wv * 8 + 8; ++kc) {
      v16bf a = Als[(fb + kc) * 32 + lane];
      v16bf b = *(const v16bf*)&vls[kc * 32 + boff];
      acc = __builtin_amdgcn_wmma_f32_16x16x32_bf16(false, a, false, b,
                                                    (short)0, acc, false, false);
    }
    if (lane == 0)       { for (int r = 0; r < 8; ++r) red[wv][r]     = acc[r]; }
    else if (lane == 16) { for (int r = 0; r < 8; ++r) red[wv][8 + r] = acc[r]; }
    __syncthreads();
    if (tid < 16) {
      float s = b_ls[wg * 16 + tid];
      for (int w2 = 0; w2 < 8; ++w2) s += red[w2][tid];
      float lat = tanhf(s);
      const int idx = 128 + wg * 16 + tid;   // latent_cat = [embV | latent | embJ]
      out[idx] = lat;                        // 1st output
      h_dec[idx] = f2bf_rne(lat);            // decoder h0 (buffer 0)
    }
  }
  gsync(bar);

  // ======================= decoder: 4094 steps ===========================
  if (wg < NT_DEC && tid < 16) cst[tid] = out[wg * 16 + tid];  // c0 = latent_cat

  for (int t = 0; t < L_SEQ - 2; ++t) {
    if (wg < NT_DEC) {
      const v16bf* Ad;
      int nkc;
      if (t == 0) {  // v = [prev0 ; latent_cat], W = [W_ih | W_hh]
        stage_async(vls, prev0, E_DIM * 2, tid);
        stage_async(vls + E_DIM, h_dec, E_DIM * 2, tid);
        Ad = Adec0; nkc = KC_DEC0;
      } else {       // prev == h  =>  gates = (W_ih + W_hh) @ h + b
        stage_async(vls, h_dec + (size_t)(t & 1) * E_DIM, E_DIM * 2, tid);
        Ad = AdecS; nkc = KC_DECS;
      }
      wait_async0();
      __syncthreads();
      v8f acc = {};
      const int per = nkc >> 1;
      const size_t fb = ((size_t)gate * NT_DEC + wg) * nkc;
      for (int kc = half * per; kc < half * per + per; ++kc) {
        v16bf a = Ad[(fb + kc) * 32 + lane];
        v16bf b = *(const v16bf*)&vls[kc * 32 + boff];
        acc = __builtin_amdgcn_wmma_f32_16x16x32_bf16(false, a, false, b,
                                                      (short)0, acc, false, false);
      }
      if (lane == 0)       { for (int r = 0; r < 8; ++r) red[wv][r]     = acc[r]; }
      else if (lane == 16) { for (int r = 0; r < 8; ++r) red[wv][8 + r] = acc[r]; }
      __syncthreads();
      if (tid < 16) {
        const int jg = wg * 16 + tid;
        float gi = red[0][tid] + red[1][tid] + b_dec[0 * E_DIM + jg];
        float gf = red[2][tid] + red[3][tid] + b_dec[1 * E_DIM + jg];
        float gg = red[4][tid] + red[5][tid] + b_dec[2 * E_DIM + jg];
        float go = red[6][tid] + red[7][tid] + b_dec[3 * E_DIM + jg];
        float c = sigm(gf) * cst[tid] + sigm(gi) * tanhf(gg);
        cst[tid] = c;
        float h = sigm(go) * tanhf(c);
        out[(size_t)E_DIM + (size_t)(L_SEQ - 2 - t) * E_DIM + jg] = h;  // recon row
        h_dec[(size_t)((t + 1) & 1) * E_DIM + jg] = f2bf_rne(h);
      }
    }
    gsync(bar);
  }
}

// ------------------------------- host side ---------------------------------

extern "C" void kernel_launch(void* const* d_in, const int* in_sizes, int n_in,
                              void* d_out, int out_size, void* d_ws, size_t ws_size,
                              hipStream_t stream) {
  (void)in_sizes; (void)n_in; (void)out_size; (void)ws_size;
  const int*   x     = (const int*)  d_in[0];
  const int*   Vg    = (const int*)  d_in[1];
  const int*   Jg    = (const int*)  d_in[2];
  /* d_in[3] = tick (unused) */
  const float* emb   = (const float*)d_in[4];
  const float* embV  = (const float*)d_in[5];
  const float* embJ  = (const float*)d_in[6];
  const float* Wih_e = (const float*)d_in[7];
  const float* Whh_e = (const float*)d_in[8];
  const float* bih_e = (const float*)d_in[9];
  const float* bhh_e = (const float*)d_in[10];
  const float* Wls   = (const float*)d_in[11];
  const float* bls   = (const float*)d_in[12];
  const float* Wih_d = (const float*)d_in[13];
  const float* Whh_d = (const float*)d_in[14];
  const float* bih_d = (const float*)d_in[15];
  const float* bhh_d = (const float*)d_in[16];
  float* out = (float*)d_out;

  char* ws = (char*)d_ws;
  size_t off = 0;
  auto take = [&](size_t bytes) -> char* {
    char* p = ws + off;
    off += (bytes + 255) & ~(size_t)255;
    return p;
  };
  unsigned short* Aenc  = (unsigned short*)take((size_t)4 * NT_ENC * KC_ENC  * 512 * 2);
  unsigned short* Als   = (unsigned short*)take((size_t)NT_LS * KC_LS        * 512 * 2);
  unsigned short* Adec0 = (unsigned short*)take((size_t)4 * NT_DEC * KC_DEC0 * 512 * 2);
  unsigned short* AdecS = (unsigned short*)take((size_t)4 * NT_DEC * KC_DECS * 512 * 2);
  unsigned short* eseq  = (unsigned short*)take((size_t)L_SEQ * E_DIM * 2);
  float*          b_enc = (float*)take((size_t)4 * H_DIM * 4);
  float*          b_dec = (float*)take((size_t)4 * E_DIM * 4);
  unsigned short* h_enc = (unsigned short*)take((size_t)2 * H_DIM * 2);
  unsigned short* h_dec = (unsigned short*)take((size_t)2 * E_DIM * 2);
  unsigned short* prev0 = (unsigned short*)take((size_t)E_DIM * 2);
  unsigned*       bar   = (unsigned*)take(256);

  k_pack_enc <<<(4 * NT_ENC * KC_ENC  * 512) / 256, 256, 0, stream>>>(Wih_e, Whh_e, Aenc);
  k_pack_ls  <<<(NT_LS * KC_LS        * 512) / 256, 256, 0, stream>>>(Wls, Als);
  k_pack_dec0<<<(4 * NT_DEC * KC_DEC0 * 512) / 256, 256, 0, stream>>>(Wih_d, Whh_d, Adec0);
  k_pack_decS<<<(4 * NT_DEC * KC_DECS * 512) / 256, 256, 0, stream>>>(Wih_d, Whh_d, AdecS);
  k_misc     <<<32, 256, 0, stream>>>(bih_e, bhh_e, bih_d, bhh_d, emb, embV, embJ,
                                      Vg, Jg, b_enc, b_dec, h_enc, h_dec, prev0, bar, out);
  k_embseq   <<<(L_SEQ * E_DIM) / 256, 256, 0, stream>>>(x, emb, eseq, out);

  k_lstm_persistent<<<NWG, 256, 0, stream>>>(
      (const v16bf*)Aenc, (const v16bf*)Als, (const v16bf*)Adec0, (const v16bf*)AdecS,
      eseq, b_enc, b_dec, bls, h_enc, h_dec, prev0, bar, out);
}
